// LorentzGNN_64003602645424
// MI455X (gfx1250) — matrix operand order repaired and checked by reading, not compiled, not run on hardware
//
#include <hip/hip_runtime.h>
#include <hip/hip_bf16.h>

#define N_NODES 32768
#define N_EDGES 524288
#define FT_IN   256
#define HID     128
#define EPSF    1e-7f

typedef __attribute__((ext_vector_type(16))) _Float16 v16h;
typedef __attribute__((ext_vector_type(8)))  float    v8f;

// ---------- helpers ----------
__device__ __forceinline__ float wave_sum(float v) {
  #pragma unroll
  for (int m = 16; m; m >>= 1) v += __shfl_xor(v, m, 32);
  return v;
}
// monotonic float<->uint mapping for atomicMax on floats
__device__ __forceinline__ unsigned enc_f(float f) {
  unsigned u = __float_as_uint(f);
  return (u & 0x80000000u) ? ~u : (u | 0x80000000u);
}
__device__ __forceinline__ float dec_f(unsigned u) {
  unsigned v = (u & 0x80000000u) ? (u & 0x7FFFFFFFu) : ~u;
  return __uint_as_float(v);
}
__device__ __forceinline__ float gelu_tanh(float x) {
  float c = 0.7978845608028654f * (x + 0.044715f * x * x * x);
  return 0.5f * x * (1.0f + tanhf(c));
}

// ---------- pack W (K x 128, f32 row-major) into WMMA B-fragment order ----------
// Layout: Wfrag[nt][kk][lane][e] with n = nt*16 + (lane&15),
//         k = kk*32 + (lane>=16 ? 16 : 0) + e    (e = 0..15, contiguous in K)
__global__ void packB_kernel(const float* __restrict__ w, _Float16* __restrict__ wfrag,
                             int K, int n_elems) {
  int i = blockIdx.x * 256 + threadIdx.x;
  if (i >= n_elems) return;
  int K32 = K >> 5;
  int e    = i & 15;
  int lane = (i >> 4) & 31;
  int t3   = i >> 9;
  int kk   = t3 % K32;
  int nt   = t3 / K32;
  int k = (kk << 5) + ((lane >> 4) << 4) + e;
  int n = (nt << 4) + (lane & 15);
  wfrag[i] = (_Float16)w[(size_t)k * HID + n];
}

// ---------- WMMA tile compute: 16 rows x 128 cols per block, 16x16 per wave ----------
// Alds holds the A tile already in per-lane fragment order:
//   Alds[((kk*32 + lane) * 16) + e]  -> one contiguous 32B read per k-step per lane
template <int K>
__device__ __forceinline__ void wmma_tile_compute(const _Float16* Alds,
                                                  const _Float16* __restrict__ Wfrag,
                                                  const float* __restrict__ bias,
                                                  float* __restrict__ out, int rowBase) {
  const int lane = threadIdx.x & 31;
  const int wave = threadIdx.x >> 5;
  const int n    = (wave << 4) + (lane & 15);   // output column
  constexpr int K32 = K / 32;

  const v16h* bfrag = (const v16h*)(Wfrag + (((size_t)wave * K32) * 32 + lane) * 16);
  const v16h* afrag = (const v16h*)(Alds + (size_t)lane * 16);

  v8f acc = {};
  #pragma unroll
  for (int kk = 0; kk < K32; ++kk) {
    v16h a = afrag[kk * 32];   // stride 32 v16h = 512 halves per k-step
    v16h b = bfrag[kk * 32];
    acc = __builtin_amdgcn_wmma_f32_16x16x32_f16(false, a, false, b, (short)0, acc,
                                                 false, false);
  }
  float bn = bias[n];
  #pragma unroll
  for (int r = 0; r < 8; ++r) {
    int row = rowBase + r + ((lane >= 16) ? 8 : 0);
    out[(size_t)row * HID + n] = acc[r] + bn;
  }
}

// A-fragment scatter index for element (row M, k):
//   rem = k & 31; lane = (rem>>4)*16 | M; elem = rem & 15
__device__ __forceinline__ int a_frag_index(int M, int k) {
  int kk = k >> 5, rem = k & 31;
  int lane_st = ((rem >> 4) << 4) | M;
  int elem = rem & 15;
  return (((kk << 5) | lane_st) << 4) | elem;
}

// ---------- layer-1 GEMM: z = logmap0(x) @ W1 + b1 ----------
__global__ void gemm1_kernel(const float* __restrict__ x, const _Float16* __restrict__ W1f,
                             const float* __restrict__ b1, float* __restrict__ z) {
  __shared__ alignas(32) _Float16 Alds[16 * FT_IN];
  __shared__ float part[16][17];
  const int t = threadIdx.x, r = t >> 4, c = t & 15;
  const int g = blockIdx.x * 16 + r;
  const float* xr = x + (size_t)g * (FT_IN + 1);
  float xs[16], ps = 0.f;
  const int k0 = c * 16;
  #pragma unroll
  for (int j = 0; j < 16; ++j) { float v = xr[1 + k0 + j]; xs[j] = v; ps += v * v; }
  part[r][c] = ps;
  __syncthreads();
  float sum = 0.f;
  #pragma unroll
  for (int j = 0; j < 16; ++j) sum += part[r][j];
  float x0    = fmaxf(xr[0], 1.0f + EPSF);
  float scale = acoshf(x0) / fmaxf(sqrtf(sum), EPSF);
  #pragma unroll
  for (int j = 0; j < 16; ++j)
    Alds[a_frag_index(r, k0 + j)] = (_Float16)(xs[j] * scale);
  __syncthreads();
  wmma_tile_compute<FT_IN>(Alds, W1f, b1, z, blockIdx.x * 16);
}

// ---------- layer-2 GEMM: z = u @ W2 + b2 (in-place safe: block-private rows) ----------
__global__ void gemm2_kernel(const float* __restrict__ u, const _Float16* __restrict__ W2f,
                             const float* __restrict__ b2, float* __restrict__ z) {
  __shared__ alignas(32) _Float16 Alds[16 * HID];
  const int t = threadIdx.x, r = t >> 4, c = t & 15;
  const int g = blockIdx.x * 16 + r;
  const float* ur = u + (size_t)g * HID;
  #pragma unroll
  for (int j = 0; j < 8; ++j)
    Alds[a_frag_index(r, c * 8 + j)] = (_Float16)ur[c * 8 + j];
  __syncthreads();
  wmma_tile_compute<HID>(Alds, W2f, b2, z, blockIdx.x * 16);
}

// ---------- attention scores + zero accumulators (wave per node) ----------
__global__ void scores_kernel(const float* __restrict__ z, const float* __restrict__ asrc,
                              const float* __restrict__ adst, float* __restrict__ s_src,
                              float* __restrict__ s_dst, float* __restrict__ agg,
                              float* __restrict__ denom, unsigned* __restrict__ mmap) {
  const int wave = threadIdx.x >> 5, lane = threadIdx.x & 31;
  const int node = blockIdx.x * 8 + wave;
  const float* zr = z + (size_t)node * HID;
  float ds = 0.f, dd = 0.f;
  #pragma unroll
  for (int j = 0; j < 4; ++j) {
    int k = lane * 4 + j;
    float v = zr[k];
    ds += v * asrc[k];
    dd += v * adst[k];
    agg[(size_t)node * HID + k] = 0.f;
  }
  ds = wave_sum(ds);
  dd = wave_sum(dd);
  if (lane == 0) { s_src[node] = ds; s_dst[node] = dd; denom[node] = 0.f; mmap[node] = 0u; }
}

// ---------- edge pass 1: segment max (thread per edge) ----------
__global__ void emax_kernel(const int* __restrict__ dst, const int* __restrict__ src,
                            const float* __restrict__ s_src, const float* __restrict__ s_dst,
                            unsigned* __restrict__ mmap) {
  const int e = blockIdx.x * 256 + threadIdx.x;
  const int d = dst[e], s = src[e];
  float v = s_dst[d] + s_src[s];
  v = (v > 0.f) ? v : 0.2f * v;
  atomicMax(&mmap[d], enc_f(v));
}

// ---------- edge pass 2: weighted scatter-add (wave per edge) ----------
__global__ void eagg_kernel(const int* __restrict__ dst, const int* __restrict__ src,
                            const float* __restrict__ s_src, const float* __restrict__ s_dst,
                            const unsigned* __restrict__ mmap, const float* __restrict__ z,
                            float* __restrict__ agg, float* __restrict__ denom) {
  const int wave = threadIdx.x >> 5, lane = threadIdx.x & 31;
  const int e = blockIdx.x * 8 + wave;
  const int d = dst[e], s = src[e];
  float v = s_dst[d] + s_src[s];
  v = (v > 0.f) ? v : 0.2f * v;
  const float w = __expf(v - dec_f(mmap[d]));
  if (lane == 0) atomicAdd(&denom[d], w);
  const float* zr = z + (size_t)s * HID;
  float* ar = agg + (size_t)d * HID;
  #pragma unroll
  for (int j = 0; j < 4; ++j) {
    int k = lane * 4 + j;
    atomicAdd(&ar[k], w * zr[k]);
  }
}

// ---------- mid pointwise: faithful expmap0 -> projx -> logmap0 -> gelu -> expmap0 -> projx -> logmap0 ----------
__global__ void mid_kernel(const float* __restrict__ agg, const float* __restrict__ denom,
                           float* __restrict__ u) {
  const int wave = threadIdx.x >> 5, lane = threadIdx.x & 31;
  const int node = blockIdx.x * 8 + wave;
  const float dn = fmaxf(denom[node], EPSF);
  float an[4], lsq = 0.f;
  #pragma unroll
  for (int j = 0; j < 4; ++j) {
    an[j] = agg[(size_t)node * HID + lane * 4 + j] / dn;
    lsq += an[j] * an[j];
  }
  float n1 = sqrtf(wave_sum(lsq));
  float coef1 = (n1 < EPSF) ? 1.f : (sinhf(n1) / n1);           // expmap0 spatial coeff
  float s1n = coef1 * n1;                                        // |spatial|
  float x0  = sqrtf(1.f + s1n * s1n);                            // projx time
  float lsc = acoshf(fmaxf(x0, 1.f + EPSF)) / fmaxf(s1n, EPSF);  // logmap0 scalar
  float g[4], gsq = 0.f;
  #pragma unroll
  for (int j = 0; j < 4; ++j) {
    g[j] = gelu_tanh(lsc * coef1 * an[j]);
    gsq += g[j] * g[j];
  }
  float n2 = sqrtf(wave_sum(gsq));
  float coef2 = (n2 < EPSF) ? 1.f : (sinhf(n2) / n2);
  float s2n = coef2 * n2;
  float x02 = sqrtf(1.f + s2n * s2n);
  float usc = acoshf(fmaxf(x02, 1.f + EPSF)) / fmaxf(s2n, EPSF) * coef2;
  #pragma unroll
  for (int j = 0; j < 4; ++j)
    u[(size_t)node * HID + lane * 4 + j] = usc * g[j];
}

// ---------- final pointwise: h = projx(expmap0(agg/den)) -> (time, spatial) ----------
__global__ void final_kernel(const float* __restrict__ agg, const float* __restrict__ denom,
                             float* __restrict__ spat, float* __restrict__ tbuf) {
  const int wave = threadIdx.x >> 5, lane = threadIdx.x & 31;
  const int node = blockIdx.x * 8 + wave;
  const float dn = fmaxf(denom[node], EPSF);
  float an[4], lsq = 0.f;
  #pragma unroll
  for (int j = 0; j < 4; ++j) {
    an[j] = agg[(size_t)node * HID + lane * 4 + j] / dn;
    lsq += an[j] * an[j];
  }
  float n = sqrtf(wave_sum(lsq));
  float coef = (n < EPSF) ? 1.f : (sinhf(n) / n);
  #pragma unroll
  for (int j = 0; j < 4; ++j)
    spat[(size_t)node * HID + lane * 4 + j] = coef * an[j];
  float sn = coef * n;
  if (lane == 0) tbuf[node] = sqrtf(1.f + sn * sn);   // projx time coordinate
}

// ---------- per-graph centroid (block per graph) ----------
__global__ void centroid_kernel(const float* __restrict__ spat, const float* __restrict__ tbuf,
                                float* __restrict__ out) {
  const int b = blockIdx.x, d = threadIdx.x;
  __shared__ float ave[130];
  __shared__ float dn;
  const int nper = N_NODES / 64;   // 512
  if (d < HID + 1) {
    float s = 0.f;
    for (int i = 0; i < nper; ++i) {
      int node = b * nper + i;
      s += (d == 0) ? tbuf[node] : spat[(size_t)node * HID + (d - 1)];
    }
    ave[d] = s / (float)nper;
  }
  __syncthreads();
  if (d == 0) {
    float inner = 0.f;
    for (int j = 1; j < HID + 1; ++j) inner += ave[j] * ave[j];
    inner -= ave[0] * ave[0];
    dn = sqrtf(fmaxf(-inner, 1e-8f));
  }
  __syncthreads();
  if (d < HID + 1) out[b * (HID + 1) + d] = ave[d] / dn;
}

// ---------- prediction head (block per graph) ----------
__global__ void head_kernel(const float* __restrict__ spat, const float* __restrict__ tbuf,
                            const float* __restrict__ Wlin, const float* __restrict__ lin_scale,
                            float* __restrict__ out) {
  const int b = blockIdx.x, j = threadIdx.x;
  __shared__ float y[HID + 1];
  __shared__ float tfac[2];
  const int node = b * (N_NODES / 64);
  if (j < HID + 1) {
    float acc = 0.f;
    for (int k = 0; k < HID + 1; ++k) {
      float gk = (k == 0) ? tbuf[node] : spat[(size_t)node * HID + (k - 1)];
      acc += gk * Wlin[k * (HID + 1) + j];
    }
    y[j] = acc;
  }
  __syncthreads();
  if (j == 0) {
    float ss = 0.f;
    for (int q = 1; q < HID + 1; ++q) ss += y[q] * y[q];
    float tt = (1.f / (1.f + __expf(-y[0]))) * lin_scale[0] + 1.1f;
    tfac[0] = tt;
    tfac[1] = sqrtf((tt * tt - 1.f) / fmaxf(ss, 1e-8f));
  }
  __syncthreads();
  if (j < HID + 1) out[b * (HID + 1) + j] = (j == 0) ? tfac[0] : y[j] * tfac[1];
}

// =====================================================================
extern "C" void kernel_launch(void* const* d_in, const int* in_sizes, int n_in,
                              void* d_out, int out_size, void* d_ws, size_t ws_size,
                              hipStream_t stream) {
  const float* x       = (const float*)d_in[0];
  const int*   eidx    = (const int*)d_in[1];
  const float* W1      = (const float*)d_in[3];
  const float* b1      = (const float*)d_in[4];
  const float* a1_src  = (const float*)d_in[5];
  const float* a1_dst  = (const float*)d_in[6];
  const float* W2      = (const float*)d_in[7];
  const float* b2      = (const float*)d_in[8];
  const float* a2_src  = (const float*)d_in[9];
  const float* a2_dst  = (const float*)d_in[10];
  const float* Wlin    = (const float*)d_in[11];
  const float* lscale  = (const float*)d_in[12];
  const int* dst = eidx;
  const int* src = eidx + N_EDGES;

  // workspace layout (256B-aligned slices)
  char* ws = (char*)d_ws;
  size_t off = 0;
  auto take = [&](size_t bytes) {
    char* p = ws + off;
    off += (bytes + 255) & ~(size_t)255;
    return p;
  };
  float*    zbuf  = (float*)take((size_t)N_NODES * HID * 4);  // z / u2 / final spatial
  float*    agg   = (float*)take((size_t)N_NODES * HID * 4);
  float*    ssrc  = (float*)take((size_t)N_NODES * 4);
  float*    sdst  = (float*)take((size_t)N_NODES * 4);
  float*    denom = (float*)take((size_t)N_NODES * 4);
  unsigned* mmap  = (unsigned*)take((size_t)N_NODES * 4);
  float*    tbuf  = (float*)take((size_t)N_NODES * 4);
  _Float16* W1f   = (_Float16*)take((size_t)FT_IN * HID * 2);
  _Float16* W2f   = (_Float16*)take((size_t)HID * HID * 2);

  float* out      = (float*)d_out;             // [64 x 129] prediction
  float* out_mean = out + 64 * (HID + 1);      // [64 x 129] graph centroid

  // weight fragment packing (one-time; L2-resident afterwards)
  packB_kernel<<<(FT_IN * HID + 255) / 256, 256, 0, stream>>>(W1, W1f, FT_IN, FT_IN * HID);
  packB_kernel<<<(HID * HID + 255) / 256, 256, 0, stream>>>(W2, W2f, HID, HID * HID);

  // ---- layer 1 ----
  gemm1_kernel<<<N_NODES / 16, 256, 0, stream>>>(x, W1f, b1, zbuf);
  scores_kernel<<<N_NODES / 8, 256, 0, stream>>>(zbuf, a1_src, a1_dst, ssrc, sdst, agg, denom, mmap);
  emax_kernel<<<N_EDGES / 256, 256, 0, stream>>>(dst, src, ssrc, sdst, mmap);
  eagg_kernel<<<N_EDGES / 8, 256, 0, stream>>>(dst, src, ssrc, sdst, mmap, zbuf, agg, denom);

  // ---- inter-layer pointwise (writes u2 over zbuf) ----
  mid_kernel<<<N_NODES / 8, 256, 0, stream>>>(agg, denom, zbuf);

  // ---- layer 2 ----
  gemm2_kernel<<<N_NODES / 16, 256, 0, stream>>>(zbuf, W2f, b2, zbuf);
  scores_kernel<<<N_NODES / 8, 256, 0, stream>>>(zbuf, a2_src, a2_dst, ssrc, sdst, agg, denom, mmap);
  emax_kernel<<<N_EDGES / 256, 256, 0, stream>>>(dst, src, ssrc, sdst, mmap);
  eagg_kernel<<<N_EDGES / 8, 256, 0, stream>>>(dst, src, ssrc, sdst, mmap, zbuf, agg, denom);

  // ---- final expmap/projx (spatial into zbuf, time into tbuf) ----
  final_kernel<<<N_NODES / 8, 256, 0, stream>>>(agg, denom, zbuf, tbuf);

  // ---- outputs ----
  centroid_kernel<<<64, 256, 0, stream>>>(zbuf, tbuf, out_mean);
  head_kernel<<<64, 256, 0, stream>>>(zbuf, tbuf, Wlin, lscale, out);
}
